// EncoderLayer_52398601011225
// MI455X (gfx1250) — matrix operand order repaired and checked by reading, not compiled, not run on hardware
//
#include <hip/hip_runtime.h>
#include <hip/hip_bf16.h>
#include <math.h>

// ---------------------------------------------------------------------------
// Transformer encoder layer for MI455X (gfx1250), bf16 WMMA everywhere.
// ---------------------------------------------------------------------------

typedef __attribute__((ext_vector_type(16))) __bf16 v16bf;
typedef __attribute__((ext_vector_type(8)))  __bf16 v8bf;
typedef __attribute__((ext_vector_type(8)))  float  v8f;

#define D_MODEL 1024
#define D_FF    4096
#define NHEADS  16
#define DHEAD   64
#define BATCH   2
#define SEQ     2048
#define ROWS    (BATCH*SEQ)

#if defined(__AMDGCN__) && defined(__gfx1250__) && \
    __has_builtin(__builtin_amdgcn_global_load_async_to_lds_b128) && \
    __has_builtin(__builtin_amdgcn_s_wait_asynccnt)
#define HAVE_ASYNC_LDS 1
#endif

#ifdef HAVE_ASYNC_LDS
// Builtin signature (probe-confirmed): arg0 = int4* in AS1, arg1 = LDS dest.
typedef __attribute__((__vector_size__(4 * sizeof(int)))) int v4i_t;
typedef __attribute__((address_space(1))) v4i_t* g_v4i;
typedef __attribute__((address_space(3))) v4i_t* l_v4i;
static __device__ __forceinline__ g_v4i as_global_v4i(const void* p) {
    return (g_v4i)(v4i_t*)const_cast<void*>(p);   // generic -> AS1
}
static __device__ __forceinline__ l_v4i as_local_v4i(void* p) {
    return (l_v4i)(v4i_t*)p;                      // generic -> AS3
}
#endif

static __device__ __forceinline__ v8bf ld8(const __bf16* p) {
    return *(const v8bf*)p;
}
static __device__ __forceinline__ v16bf cat16(v8bf a, v8bf b) {
    return __builtin_shufflevector(a, b, 0,1,2,3,4,5,6,7,8,9,10,11,12,13,14,15);
}

// A-fragment: 16x32 bf16 (M x K). ISA layout: lane m = lane&15, lane-half
// selects K-offset 0/8; VGPR0-3 hold K {kh..kh+7}, VGPR4-7 hold {16+kh..}.
// Both chunks are 16B-contiguous in row-major A -> two b128 loads.
static __device__ __forceinline__ v16bf loadA(const __bf16* __restrict__ A,
                                              int lda, int row0, int k0, int lane) {
    int m  = row0 + (lane & 15);
    int kh = (lane >> 4) << 3;
    const __bf16* p = A + (size_t)m * lda + k0;
    return cat16(ld8(p + kh), ld8(p + 16 + kh));
}

// B-fragment: 32x16 bf16 (K x N), sourced from Bt stored [N,K] row-major.
// Lane n = lane&15, lane-half selects K-offset 0/8; K {kb..kb+7}, {16+kb..} -> two b128 loads.
// Works for both global and LDS source pointers.
static __device__ __forceinline__ v16bf loadB(const __bf16* Bt,
                                              int ldb, int n0, int k0, int lane) {
    int n  = n0 + (lane & 15);
    int kb = (lane >> 4) << 3;
    const __bf16* p = Bt + (size_t)n * ldb + k0 + kb;
    return cat16(ld8(p), ld8(p + 16));
}

// ---------------------------------------------------------------------------
// Elementwise helpers
// ---------------------------------------------------------------------------
__global__ void cvt_bf16_kernel(const float* __restrict__ in,
                                __bf16* __restrict__ out, size_t n) {
    size_t i = (size_t)blockIdx.x * blockDim.x + threadIdx.x;
    if (i < n) out[i] = (__bf16)in[i];
}

// Wt[n*K + k] = (bf16) W[k*N + n]   (weights [K,N] f32 -> [N,K] bf16)
__global__ void transcvt_kernel(const float* __restrict__ W,
                                __bf16* __restrict__ Wt, int K, int N) {
    size_t i = (size_t)blockIdx.x * blockDim.x + threadIdx.x;
    if (i < (size_t)K * N) {
        int k = (int)(i % K);
        int n = (int)(i / K);
        Wt[i] = (__bf16)W[(size_t)k * N + n];
    }
}

// Vt[(b*H+h)*64 + d][s] = V[(b*S+s)*1024 + h*64 + d]
__global__ void vtrans_kernel(const __bf16* __restrict__ V,
                              __bf16* __restrict__ Vt) {
    size_t total = (size_t)BATCH * NHEADS * DHEAD * SEQ;
    size_t i = (size_t)blockIdx.x * blockDim.x + threadIdx.x;
    if (i < total) {
        int s = (int)(i % SEQ);
        size_t t = i / SEQ;
        int d = (int)(t % DHEAD);
        size_t bh = t / DHEAD;
        int h = (int)(bh % NHEADS);
        int b = (int)(bh / NHEADS);
        Vt[i] = V[((size_t)b * SEQ + s) * D_MODEL + h * DHEAD + d];
    }
}

// ---------------------------------------------------------------------------
// Generic bf16 WMMA GEMM: C = A[M,K] @ Bt[N,K]^T + bias (+resid)(gelu?)
// 256 threads = 8 waves; block tile 256x128; wave tile 64x64 (4x4 WMMA tiles,
// 16 f32 accumulators = 128 VGPRs). All 8 A-frag + 8 B-frag b128 loads are
// issued up front each K-step so waits overlap the WMMA block. Epilogue
// variants are compile-time template parameters (branchless).
// ---------------------------------------------------------------------------
template<bool HAS_RESID, bool DO_GELU, bool OUT_F32>
__global__ __launch_bounds__(256)
void gemm_bf16_kernel(const __bf16* __restrict__ A, const __bf16* __restrict__ Bt,
                      const float* __restrict__ bias, const float* __restrict__ resid,
                      float* __restrict__ outF, __bf16* __restrict__ outB,
                      int M, int N, int K) {
    int lane = threadIdx.x & 31;
    int w    = threadIdx.x >> 5;
    int wm   = w & 3;            // 0..3
    int wn   = w >> 2;           // 0..1
    int row0 = blockIdx.y * 256 + wm * 64;
    int col0 = blockIdx.x * 128 + wn * 64;

    v8f acc[4][4] = {};

    for (int k0 = 0; k0 < K; k0 += 32) {
        v16bf a[4];
        v16bf bfr[4];
#pragma unroll
        for (int mt = 0; mt < 4; ++mt)
            a[mt] = loadA(A, K, row0 + mt * 16, k0, lane);
#pragma unroll
        for (int nt = 0; nt < 4; ++nt)
            bfr[nt] = loadB(Bt, K, col0 + nt * 16, k0, lane);
#pragma unroll
        for (int nt = 0; nt < 4; ++nt)
#pragma unroll
            for (int mt = 0; mt < 4; ++mt)
                acc[mt][nt] = __builtin_amdgcn_wmma_f32_16x16x32_bf16(
                    false, a[mt], false, bfr[nt], (short)0, acc[mt][nt], false, false);
    }

    int hi = lane >> 4;
    int nl = lane & 15;
#pragma unroll
    for (int mt = 0; mt < 4; ++mt) {
#pragma unroll
        for (int nt = 0; nt < 4; ++nt) {
#pragma unroll
            for (int r = 0; r < 8; ++r) {
                int row = row0 + mt * 16 + r + hi * 8;
                int col = col0 + nt * 16 + nl;
                float v = acc[mt][nt][r] + bias[col];
                if constexpr (HAS_RESID) v += resid[(size_t)row * N + col];
                if constexpr (DO_GELU)
                    v = 0.5f * v * (1.0f + erff(v * 0.70710678118654752f));
                if constexpr (OUT_F32) outF[(size_t)row * N + col] = v;
                else                   outB[(size_t)row * N + col] = (__bf16)v;
            }
        }
    }
}

// ---------------------------------------------------------------------------
// Flash attention: per block one (b,h) and a 128-query tile; 8 waves, each
// owning 16 query rows. K/V tiles staged block-wide in LDS via async-to-LDS
// (ASYNCcnt) when available; online softmax in f32 registers; P staged
// per-wave in LDS (same-wave DS ops in-order -> no barrier) and re-read in
// A-fragment layout for the P@V WMMAs.
// ---------------------------------------------------------------------------
__global__ __launch_bounds__(256)
void attn_kernel(const __bf16* __restrict__ Q, const __bf16* __restrict__ Km,
                 const __bf16* __restrict__ Vt, const unsigned char* __restrict__ mask,
                 __bf16* __restrict__ Ctx) {
    __shared__ __bf16 Pbuf[8 * 16 * 128];     // 32 KB: per-wave 16x128 P tiles
    __shared__ __bf16 KLds[128 * 64];         // 16 KB: key block  [s_local][d]
    __shared__ __bf16 VLds[64 * 128];         // 16 KB: value blk  [d][s_local]

    int lane = threadIdx.x & 31;
    int w    = threadIdx.x >> 5;
    int bh   = blockIdx.y;
    int b    = bh / NHEADS;
    int h    = bh % NHEADS;
    int qrow0 = blockIdx.x * 128 + w * 16;    // sequence-local

    const __bf16* Qbase = Q  + (size_t)(b * SEQ) * D_MODEL + h * DHEAD;
    const __bf16* Kbase = Km + (size_t)(b * SEQ) * D_MODEL + h * DHEAD;
    const __bf16* Vbase = Vt + (size_t)bh * DHEAD * SEQ;
    __bf16* Pw = &Pbuf[w * 16 * 128];

    // Q fragments for this wave's 16 rows, K=64 -> 2 fragments, kept in regs.
    v16bf aq0 = loadA(Qbase, D_MODEL, qrow0, 0,  lane);
    v16bf aq1 = loadA(Qbase, D_MODEL, qrow0, 32, lane);

    float mrow[8], lrow[8];
#pragma unroll
    for (int r = 0; r < 8; ++r) { mrow[r] = -1e30f; lrow[r] = 0.0f; }
    v8f o[4] = {};

    const float scale = 0.125f;   // 1/sqrt(64)
    int hi = lane >> 4;
    int nl = lane & 15;

    for (int j = 0; j < SEQ; j += 128) {
        // ---- stage K (128x64) and V (64x128) tiles into LDS, block-wide ----
#ifdef HAVE_ASYNC_LDS
#pragma unroll
        for (int t = 0; t < 4; ++t) {
            int c = (int)threadIdx.x + t * 256;         // 1024 chunks of 16B
            int kr = c >> 3, kc = (c & 7) << 3;         // K tile: 8 chunks/row
            __builtin_amdgcn_global_load_async_to_lds_b128(
                as_global_v4i(Kbase + (size_t)(j + kr) * D_MODEL + kc),
                as_local_v4i(&KLds[kr * 64 + kc]), 0, 0);
            int vr = c >> 4, vc = (c & 15) << 3;        // V tile: 16 chunks/row
            __builtin_amdgcn_global_load_async_to_lds_b128(
                as_global_v4i(Vbase + (size_t)vr * SEQ + j + vc),
                as_local_v4i(&VLds[vr * 128 + vc]), 0, 0);
        }
        __builtin_amdgcn_s_wait_asynccnt(0);
        __syncthreads();
#else
#pragma unroll
        for (int t = 0; t < 4; ++t) {
            int c = (int)threadIdx.x + t * 256;
            int kr = c >> 3, kc = (c & 7) << 3;
            *(v8bf*)&KLds[kr * 64 + kc] =
                ld8(Kbase + (size_t)(j + kr) * D_MODEL + kc);
            int vr = c >> 4, vc = (c & 15) << 3;
            *(v8bf*)&VLds[vr * 128 + vc] =
                ld8(Vbase + (size_t)vr * SEQ + j + vc);
        }
        __syncthreads();
#endif

        // scores S = Q K^T for 16 q-rows x 128 keys (8 WMMA n-tiles, 2 k-steps)
        v8f s[8] = {};
#pragma unroll
        for (int nt = 0; nt < 8; ++nt) {
            v16bf bk0 = loadB(KLds, DHEAD, nt * 16, 0,  lane);
            s[nt] = __builtin_amdgcn_wmma_f32_16x16x32_bf16(
                false, aq0, false, bk0, (short)0, s[nt], false, false);
            v16bf bk1 = loadB(KLds, DHEAD, nt * 16, 32, lane);
            s[nt] = __builtin_amdgcn_wmma_f32_16x16x32_bf16(
                false, aq1, false, bk1, (short)0, s[nt], false, false);
        }

        // online softmax per accumulator row (row = r + 8*hi, matching C/D map)
#pragma unroll
        for (int r = 0; r < 8; ++r) {
            int qrow = qrow0 + r + hi * 8;
            const unsigned char* mp = mask + ((size_t)b * SEQ + qrow) * SEQ + j + nl;
            float tmax = -1e30f;
#pragma unroll
            for (int nt = 0; nt < 8; ++nt) {
                float v = s[nt][r] * scale;
                if (mp[nt * 16]) v = -1e9f;
                s[nt][r] = v;
                tmax = fmaxf(tmax, v);
            }
            for (int off = 1; off < 16; off <<= 1)
                tmax = fmaxf(tmax, __shfl_xor(tmax, off, 16));
            float mnew = fmaxf(mrow[r], tmax);
            float resc = __expf(mrow[r] - mnew);
            mrow[r] = mnew;
            float rsum = 0.0f;
#pragma unroll
            for (int nt = 0; nt < 8; ++nt) {
                float p = __expf(s[nt][r] - mnew);
                rsum += p;
                Pw[(r + hi * 8) * 128 + nt * 16 + nl] = (__bf16)p;
            }
            for (int off = 1; off < 16; off <<= 1)
                rsum += __shfl_xor(rsum, off, 16);
            lrow[r] = lrow[r] * resc + rsum;
#pragma unroll
            for (int ot = 0; ot < 4; ++ot) o[ot][r] *= resc;
        }

        // O += P @ V_blk : P from LDS as A-fragments (4 k-steps of 32 keys),
        // V from LDS ([d, s_local] row-major) as B-fragments.
#pragma unroll
        for (int kk = 0; kk < 4; ++kk) {
            int m  = lane & 15;
            int kh = (lane >> 4) << 3;
            const __bf16* pp = Pw + m * 128 + kk * 32;
            v16bf pa = cat16(*(const v8bf*)(pp + kh), *(const v8bf*)(pp + 16 + kh));
#pragma unroll
            for (int ot = 0; ot < 4; ++ot) {
                v16bf bv = loadB(VLds, 128, ot * 16, kk * 32, lane);
                o[ot] = __builtin_amdgcn_wmma_f32_16x16x32_bf16(
                    false, pa, false, bv, (short)0, o[ot], false, false);
            }
        }
        __syncthreads();   // before next iteration overwrites K/V tiles
    }

    // normalize and write ctx (bf16, [B*S, H*64] layout)
#pragma unroll
    for (int r = 0; r < 8; ++r) {
        float inv = 1.0f / lrow[r];
        int qrow = qrow0 + r + hi * 8;
#pragma unroll
        for (int ot = 0; ot < 4; ++ot) {
            Ctx[((size_t)b * SEQ + qrow) * D_MODEL + h * DHEAD + ot * 16 + nl] =
                (__bf16)(o[ot][r] * inv);
        }
    }
}

// ---------------------------------------------------------------------------
// Row LayerNorm over D_MODEL; writes f32 and optionally bf16.
// ---------------------------------------------------------------------------
__global__ __launch_bounds__(256)
void layernorm_kernel(const float* __restrict__ X, const float* __restrict__ g,
                      const float* __restrict__ bta, float* __restrict__ outF,
                      __bf16* __restrict__ outB) {
    __shared__ float sbuf[256];
    __shared__ float ssbuf[256];
    int row = blockIdx.x;
    const float* xr = X + (size_t)row * D_MODEL;
    float s = 0.0f, ss = 0.0f;
    for (int i = threadIdx.x; i < D_MODEL; i += 256) {
        float v = xr[i];
        s += v; ss += v * v;
    }
    sbuf[threadIdx.x] = s; ssbuf[threadIdx.x] = ss;
    __syncthreads();
    for (int off = 128; off > 0; off >>= 1) {
        if (threadIdx.x < off) {
            sbuf[threadIdx.x]  += sbuf[threadIdx.x + off];
            ssbuf[threadIdx.x] += ssbuf[threadIdx.x + off];
        }
        __syncthreads();
    }
    float mu  = sbuf[0] * (1.0f / D_MODEL);
    float var = ssbuf[0] * (1.0f / D_MODEL) - mu * mu;
    float rs  = rsqrtf(var + 1e-5f);
    for (int i = threadIdx.x; i < D_MODEL; i += 256) {
        float y = (xr[i] - mu) * rs * g[i] + bta[i];
        if (outF) outF[(size_t)row * D_MODEL + i] = y;
        if (outB) outB[(size_t)row * D_MODEL + i] = (__bf16)y;
    }
}

// ---------------------------------------------------------------------------
// Launch
// ---------------------------------------------------------------------------
extern "C" void kernel_launch(void* const* d_in, const int* in_sizes, int n_in,
                              void* d_out, int out_size, void* d_ws, size_t ws_size,
                              hipStream_t stream) {
    (void)in_sizes; (void)n_in; (void)out_size; (void)ws_size;
    const float* X   = (const float*)d_in[0];
    const unsigned char* mask = (const unsigned char*)d_in[1];
    const float* Wq  = (const float*)d_in[2];
    const float* bq  = (const float*)d_in[3];
    const float* Wk  = (const float*)d_in[4];
    const float* bk  = (const float*)d_in[5];
    const float* Wv  = (const float*)d_in[6];
    const float* bv  = (const float*)d_in[7];
    const float* Wo  = (const float*)d_in[8];
    const float* bo  = (const float*)d_in[9];
    const float* g1  = (const float*)d_in[10];
    const float* b1  = (const float*)d_in[11];
    const float* W1  = (const float*)d_in[12];
    const float* bf1 = (const float*)d_in[13];
    const float* W2  = (const float*)d_in[14];
    const float* bf2 = (const float*)d_in[15];
    const float* g2  = (const float*)d_in[16];
    const float* b2  = (const float*)d_in[17];
    float* out = (float*)d_out;

    char* ws = (char*)d_ws;
    size_t off = 0;
    auto alloc = [&](size_t bytes) -> char* {
        char* p = ws + off;
        off += (bytes + 255) & ~(size_t)255;
        return p;
    };
    __bf16* Xb    = (__bf16*)alloc((size_t)ROWS * D_MODEL * 2);
    __bf16* Wqt   = (__bf16*)alloc((size_t)D_MODEL * D_MODEL * 2);
    __bf16* Wkt   = (__bf16*)alloc((size_t)D_MODEL * D_MODEL * 2);
    __bf16* Wvt   = (__bf16*)alloc((size_t)D_MODEL * D_MODEL * 2);
    __bf16* Wot   = (__bf16*)alloc((size_t)D_MODEL * D_MODEL * 2);
    __bf16* W1t   = (__bf16*)alloc((size_t)D_MODEL * D_FF * 2);
    __bf16* W2t   = (__bf16*)alloc((size_t)D_FF * D_MODEL * 2);
    __bf16* Qb    = (__bf16*)alloc((size_t)ROWS * D_MODEL * 2);
    __bf16* Kb    = (__bf16*)alloc((size_t)ROWS * D_MODEL * 2);
    __bf16* Vb    = (__bf16*)alloc((size_t)ROWS * D_MODEL * 2);
    __bf16* Vt    = (__bf16*)alloc((size_t)BATCH * NHEADS * DHEAD * SEQ * 2);
    __bf16* Ctx   = (__bf16*)alloc((size_t)ROWS * D_MODEL * 2);
    float*  res1  = (float*)alloc((size_t)ROWS * D_MODEL * 4);
    float*  attnF = (float*)alloc((size_t)ROWS * D_MODEL * 4);
    __bf16* attnB = (__bf16*)alloc((size_t)ROWS * D_MODEL * 2);
    __bf16* Hb    = (__bf16*)alloc((size_t)ROWS * D_FF * 2);
    float*  res2  = (float*)alloc((size_t)ROWS * D_MODEL * 4);

    const size_t nX = (size_t)ROWS * D_MODEL;
    cvt_bf16_kernel<<<(nX + 255) / 256, 256, 0, stream>>>(X, Xb, nX);

    const size_t nW = (size_t)D_MODEL * D_MODEL;
    transcvt_kernel<<<(nW + 255) / 256, 256, 0, stream>>>(Wq, Wqt, D_MODEL, D_MODEL);
    transcvt_kernel<<<(nW + 255) / 256, 256, 0, stream>>>(Wk, Wkt, D_MODEL, D_MODEL);
    transcvt_kernel<<<(nW + 255) / 256, 256, 0, stream>>>(Wv, Wvt, D_MODEL, D_MODEL);
    transcvt_kernel<<<(nW + 255) / 256, 256, 0, stream>>>(Wo, Wot, D_MODEL, D_MODEL);
    const size_t nW1 = (size_t)D_MODEL * D_FF;
    transcvt_kernel<<<(nW1 + 255) / 256, 256, 0, stream>>>(W1, W1t, D_MODEL, D_FF);
    transcvt_kernel<<<(nW1 + 255) / 256, 256, 0, stream>>>(W2, W2t, D_FF, D_MODEL);

    // Q, K, V projections (bias, bf16 out)
    dim3 gProj(D_MODEL / 128, ROWS / 256);
    gemm_bf16_kernel<false, false, false><<<gProj, 256, 0, stream>>>(
        Xb, Wqt, bq, nullptr, nullptr, Qb, ROWS, D_MODEL, D_MODEL);
    gemm_bf16_kernel<false, false, false><<<gProj, 256, 0, stream>>>(
        Xb, Wkt, bk, nullptr, nullptr, Kb, ROWS, D_MODEL, D_MODEL);
    gemm_bf16_kernel<false, false, false><<<gProj, 256, 0, stream>>>(
        Xb, Wvt, bv, nullptr, nullptr, Vb, ROWS, D_MODEL, D_MODEL);

    const size_t nV = (size_t)BATCH * NHEADS * DHEAD * SEQ;
    vtrans_kernel<<<(nV + 255) / 256, 256, 0, stream>>>(Vb, Vt);

    // flash attention -> Ctx
    dim3 gAttn(SEQ / 128, BATCH * NHEADS);
    attn_kernel<<<gAttn, 256, 0, stream>>>(Qb, Kb, Vt, mask, Ctx);

    // output projection + residual(X) -> res1 (f32)
    gemm_bf16_kernel<true, false, true><<<gProj, 256, 0, stream>>>(
        Ctx, Wot, bo, X, res1, nullptr, ROWS, D_MODEL, D_MODEL);
    // LN1 -> attnF (f32) + attnB (bf16)
    layernorm_kernel<<<ROWS, 256, 0, stream>>>(res1, g1, b1, attnF, attnB);

    // FFN up: H = gelu(attn @ W1 + bf1), bf16 out
    dim3 gFF1(D_FF / 128, ROWS / 256);
    gemm_bf16_kernel<false, true, false><<<gFF1, 256, 0, stream>>>(
        attnB, W1t, bf1, nullptr, nullptr, Hb, ROWS, D_FF, D_MODEL);
    // FFN down + residual(attnF) -> res2 (f32)
    gemm_bf16_kernel<true, false, true><<<gProj, 256, 0, stream>>>(
        Hb, W2t, bf2, attnF, res2, nullptr, ROWS, D_MODEL, D_FF);
    // LN2 -> final f32 output
    layernorm_kernel<<<ROWS, 256, 0, stream>>>(res2, g2, b2, out, nullptr);
}
